// PointTransformerFromCoords_47278999994581
// MI455X (gfx1250) — matrix operand order repaired and checked by reading, not compiled, not run on hardware
//
#include <hip/hip_runtime.h>
#include <hip/hip_bf16.h>
#include <math.h>

// ---------------------------------------------------------------------------
// Point Transformer (vector attention) fully fused for gfx1250 / MI455X.
//  Kernel 1: coord-embed + q/k/v projections (tiny, scalar).
//  Kernel 2: per (batch, 16-query tile) block; loops over 16-key tiles;
//            pos-MLP, attn-MLP run on v_wmma_f32_16x16x32_f16; online
//            per-channel softmax + aggregation in fp32; fused output linear.
//  k/v tiles are double-buffered in LDS via GLOBAL_LOAD_ASYNC_TO_LDS_B128
//  (ASYNCcnt) when the toolchain exposes the builtin; sync fallback otherwise.
// ---------------------------------------------------------------------------

typedef __attribute__((ext_vector_type(16))) _Float16 v16h;
typedef __attribute__((ext_vector_type(8)))  float    v8f;

#define WMMA_F16(a, b, c) \
  __builtin_amdgcn_wmma_f32_16x16x32_f16(false, (a), false, (b), (short)0, (c), false, false)

// DS ops from one wave execute in order; this waits them out and stops the
// compiler from reordering LDS traffic across the staging hop.
#define WAVE_LDS_FENCE() asm volatile("s_wait_dscnt 0" ::: "memory")

static constexpr int B_  = 2;
static constexpr int N_  = 768;
static constexpr int C_  = 64;
static constexpr int PH_ = 32;
static constexpr int AH_ = 32;
static constexpr int BN_ = B_ * N_;
static constexpr int JTILES = N_ / 16;   // 48
static constexpr float NEG_ = -1e9f;

// ---------------- async global -> LDS staging helpers ----------------------

typedef int vint4_t __attribute__((vector_size(4 * sizeof(int))));
#define AS_GLOBAL __attribute__((address_space(1)))
#define AS_LOCAL  __attribute__((address_space(3)))

// Copy 16 bytes from global to LDS. Prefers the CDNA5 async-DMA path
// (global_load_async_to_lds_b128, tracked with ASYNCcnt); falls back to a
// synchronous VGPR-mediated copy issued at the same point in the schedule.
__device__ inline void async_copy16B(const float* gsrc, float* ldst) {
#if __has_builtin(__builtin_amdgcn_global_load_async_to_lds_b128)
  AS_GLOBAL void* g = (AS_GLOBAL void*)const_cast<float*>(gsrc);
  AS_LOCAL  void* l = (AS_LOCAL void*)ldst;
  __builtin_amdgcn_global_load_async_to_lds_b128(
      (AS_GLOBAL vint4_t*)g, (AS_LOCAL vint4_t*)l, 0 /*offset*/, 0 /*cpol*/);
#else
  *(float4*)ldst = *(const float4*)gsrc;
#endif
}

__device__ inline void wait_async() {
#if __has_builtin(__builtin_amdgcn_s_wait_asynccnt)
  __builtin_amdgcn_s_wait_asynccnt(0);
#else
  asm volatile("s_wait_asynccnt 0" ::: "memory");
#endif
}

// ---------------- fragment helpers (ISA 7.12.2 wave32 layouts) -------------

// A fragment (16x32, f16): lane L holds row (L&15); lanes<16 carry K
// [0..7]+[16..23], lanes>=16 carry K [8..15]+[24..31]. Staging tile is
// row-major [16][64] f16 in LDS -> two 16B ds_load_b128 per fragment.
__device__ inline v16h load_a_frag(const _Float16* st, int kbase) {
  int lane = threadIdx.x & 31;
  int row  = lane & 15;
  int koff = (lane & 16) ? 8 : 0;
  const _Float16* p = st + row * 64 + kbase + koff;
  v16h a;
  ((float4*)&a)[0] = *(const float4*)(p);
  ((float4*)&a)[1] = *(const float4*)(p + 16);
  return a;
}

// B fragment (32x16, f16) from row-major fp32 W[K][Nfull], K-chunk kb,
// N-tile nt: lane L holds col (L&15); lanes<16 K=kb+[0..7]+[16..23], lanes>=16 K=kb+[8..15]+[24..31].
__device__ inline v16h load_b_frag(const float* W, int Nfull, int kb, int nt, int Kvalid) {
  int lane = threadIdx.x & 31;
  int col  = nt * 16 + (lane & 15);
  int k0   = kb + ((lane & 16) ? 8 : 0);
  v16h b;
#pragma unroll
  for (int e = 0; e < 16; ++e) {
    int k = k0 + (e < 8 ? e : e + 8);
    float w = (k < Kvalid) ? W[k * Nfull + col] : 0.0f;
    b[e] = (_Float16)w;
  }
  return b;
}

// C/D layout: VGPR r holds (M = r + (lane<16?0:8), N = lane&15).
__device__ inline v8f bcast_bias(const float* bias, int nt) {
  int lane = threadIdx.x & 31;
  float v = bias[nt * 16 + (lane & 15)];
  v8f c;
#pragma unroll
  for (int r = 0; r < 8; ++r) c[r] = v;
  return c;
}

__device__ inline v8f relu8(v8f x) {
  v8f r;
#pragma unroll
  for (int r_ = 0; r_ < 8; ++r_) r[r_] = fmaxf(x[r_], 0.0f);
  return r;
}

// Store a D fragment into the row-major f16 staging tile so it can be
// re-read as an A fragment (rows = M, cols = nbase + N).
__device__ inline void store_d_rows(v8f d, _Float16* st, int nbase) {
  int lane = threadIdx.x & 31;
  int col  = nbase + (lane & 15);
  int mo   = (lane & 16) ? 8 : 0;
#pragma unroll
  for (int r = 0; r < 8; ++r) st[(mo + r) * 64 + col] = (_Float16)d[r];
}

// lane <-> lane^16 exchange (SWAPX16 ds_swizzle, offset = xor 0x10, and 0x1f)
__device__ inline float swap16(float x) {
  int i = __builtin_amdgcn_ds_swizzle(__float_as_int(x), 0x401F);
  return __int_as_float(i);
}

// ------------------------- kernel 1: embed + qkv ---------------------------

__global__ void pt_embed_qkv(const float* __restrict__ coords,
                             const float* __restrict__ ce_w1, const float* __restrict__ ce_b1,
                             const float* __restrict__ ce_w2, const float* __restrict__ ce_b2,
                             const float* __restrict__ wq, const float* __restrict__ wk,
                             const float* __restrict__ wv,
                             float* __restrict__ xg, float* __restrict__ qg,
                             float* __restrict__ kg, float* __restrict__ vg) {
  int p = blockIdx.x;          // point index in [0, B*N)
  int t = threadIdx.x;         // channel, blockDim = 64
  __shared__ float c3[3];
  __shared__ float hs[C_];
  __shared__ float xs[C_];
  if (t < 3) c3[t] = coords[p * 3 + t];
  __syncthreads();
  float h = ce_b1[t];
#pragma unroll
  for (int d = 0; d < 3; ++d) h += c3[d] * ce_w1[d * C_ + t];
  hs[t] = fmaxf(h, 0.0f);
  __syncthreads();
  float xv = ce_b2[t];
  for (int k = 0; k < C_; ++k) xv += hs[k] * ce_w2[k * C_ + t];
  xs[t] = xv;
  xg[p * C_ + t] = xv;
  __syncthreads();
  float qv = 0.f, kv = 0.f, vv = 0.f;
  for (int k = 0; k < C_; ++k) {
    float xk = xs[k];
    qv += xk * wq[k * C_ + t];
    kv += xk * wk[k * C_ + t];
    vv += xk * wv[k * C_ + t];
  }
  qg[p * C_ + t] = qv;
  kg[p * C_ + t] = kv;
  vg[p * C_ + t] = vv;
}

// --------------------- kernel 2: fused attention core ----------------------

__global__ __launch_bounds__(256)
void pt_attention(const float* __restrict__ coords, const int* __restrict__ maskp,
                  const float* __restrict__ qg, const float* __restrict__ kg,
                  const float* __restrict__ vg, const float* __restrict__ xg,
                  const float* __restrict__ pm_w1, const float* __restrict__ pm_b1,
                  const float* __restrict__ pm_w2, const float* __restrict__ pm_b2,
                  const float* __restrict__ am_w1, const float* __restrict__ am_b1,
                  const float* __restrict__ am_w2, const float* __restrict__ am_b2,
                  const float* __restrict__ out_w, const float* __restrict__ out_b,
                  float* __restrict__ outp) {
  const int b     = blockIdx.x / JTILES;   // grid = B * 48
  const int itile = blockIdx.x % JTILES;
  const int ibase = itile * 16;
  const int tid   = threadIdx.x;
  const int wave  = tid >> 5;
  const int lane  = tid & 31;
  const int ch_lo = lane & 15;
  const int mo    = (lane & 16) ? 8 : 0;

  __shared__ float q_s[16 * C_];
  __shared__ __align__(16) float k_buf[2][16 * C_];   // double-buffered (async DMA)
  __shared__ __align__(16) float v_buf[2][16 * C_];
  __shared__ float outpre_s[16 * C_];
  __shared__ float ci_s[16 * 3];
  __shared__ float cj_s[16 * 3];
  __shared__ int   mask_s[16];
  __shared__ __align__(16) _Float16 stage[8 * 16 * C_];   // per-wave 16x64 f16
  _Float16* st = stage + wave * 16 * C_;

  // ---- kick off async DMA of key/value tile 0 (each thread moves 16B) ----
  async_copy16B(kg + (size_t)(b * N_) * C_ + tid * 4, &k_buf[0][tid * 4]);
  async_copy16B(vg + (size_t)(b * N_) * C_ + tid * 4, &v_buf[0][tid * 4]);

  // ---- load query tile (q + coords) once ----
  for (int idx = tid; idx < 16 * C_; idx += 256)
    q_s[idx] = qg[(b * N_ + ibase) * C_ + idx];
  if (tid < 48) ci_s[tid] = coords[(b * N_ + ibase) * 3 + tid];

  // ---- weight fragments (per wave, held in VGPRs) ----
  v16h pmw1[2], pmw2[4], amw1[2][2], amw2[4];
#pragma unroll
  for (int nt = 0; nt < 2; ++nt) pmw1[nt] = load_b_frag(pm_w1, PH_, 0, nt, 3);
#pragma unroll
  for (int nt = 0; nt < 4; ++nt) pmw2[nt] = load_b_frag(pm_w2, C_, 0, nt, PH_);
#pragma unroll
  for (int kc = 0; kc < 2; ++kc)
#pragma unroll
    for (int nt = 0; nt < 2; ++nt) amw1[kc][nt] = load_b_frag(am_w1, AH_, kc * 32, nt, C_);
#pragma unroll
  for (int nt = 0; nt < 4; ++nt) amw2[nt] = load_b_frag(am_w2, C_, 0, nt, AH_);

  // ---- online softmax state: per lane, per owned query, per channel tile ----
  float mrun[2][4], srun[2][4], arun[2][4];
#pragma unroll
  for (int ii = 0; ii < 2; ++ii)
#pragma unroll
    for (int nt = 0; nt < 4; ++nt) { mrun[ii][nt] = -3.0e38f; srun[ii][nt] = 0.f; arun[ii][nt] = 0.f; }

  for (int jt = 0; jt < JTILES; ++jt) {
    const int jbase = jt * 16;
    const int cur   = jt & 1;

    // current tile's DMA complete (per-wave), then block-wide visibility +
    // guarantee the other buffer is no longer being read.
    wait_async();
    __syncthreads();

    // issue DMA for the next tile into the other buffer
    if (jt + 1 < JTILES) {
      const size_t nb = (size_t)(b * N_ + jbase + 16) * C_;
      async_copy16B(kg + nb + tid * 4, &k_buf[1 - cur][tid * 4]);
      async_copy16B(vg + nb + tid * 4, &v_buf[1 - cur][tid * 4]);
    }
    // prefetch the tile after that into L2 (global_prefetch_b8)
    if (jt + 2 < JTILES) {
      const size_t pb = (size_t)(b * N_ + jbase + 32) * C_;
      __builtin_prefetch(kg + pb + tid * 4, 0, 1);
      __builtin_prefetch(vg + pb + tid * 4, 0, 1);
    }

    if (tid < 48) cj_s[tid] = coords[(b * N_ + jbase) * 3 + tid];
    if (tid < 16) mask_s[tid] = maskp[b * N_ + jbase + tid];
    __syncthreads();

    const float* k_s = k_buf[cur];
    const float* v_s = v_buf[cur];

#pragma unroll
    for (int ii = 0; ii < 2; ++ii) {
      const int i_loc = wave * 2 + ii;

      // ---- A fragment of rel_pos (16 pair-rows x K=3, zero padded) ----
      int row = lane & 15;
      bool lo = (lane & 16) == 0;
      float rx = lo ? (ci_s[i_loc * 3 + 0] - cj_s[row * 3 + 0]) : 0.f;
      float ry = lo ? (ci_s[i_loc * 3 + 1] - cj_s[row * 3 + 1]) : 0.f;
      float rz = lo ? (ci_s[i_loc * 3 + 2] - cj_s[row * 3 + 2]) : 0.f;
      v16h arel;
#pragma unroll
      for (int e = 0; e < 16; ++e) arel[e] = (_Float16)0.0f;
      arel[0] = (_Float16)rx; arel[1] = (_Float16)ry; arel[2] = (_Float16)rz;

      // ---- pos MLP layer 1: (16x3)@(3x32) + b, relu ----
      v8f h0 = relu8(WMMA_F16(arel, pmw1[0], bcast_bias(pm_b1, 0)));
      v8f h1 = relu8(WMMA_F16(arel, pmw1[1], bcast_bias(pm_b1, 1)));

      WAVE_LDS_FENCE();
      store_d_rows(h0, st, 0);
      store_d_rows(h1, st, 16);
      WAVE_LDS_FENCE();
      v16h ah = load_a_frag(st, 0);

      // ---- pos MLP layer 2 -> pos_enc fp32 fragments ----
      v8f pe[4];
#pragma unroll
      for (int nt = 0; nt < 4; ++nt)
        pe[nt] = WMMA_F16(ah, pmw2[nt], bcast_bias(pm_b2, nt));

      // ---- attn input a = q_i - k_j + pos_enc, stage to f16 16x64 ----
      WAVE_LDS_FENCE();
#pragma unroll
      for (int nt = 0; nt < 4; ++nt) {
        int ch = nt * 16 + ch_lo;
        float qv = q_s[i_loc * C_ + ch];
#pragma unroll
        for (int r = 0; r < 8; ++r) {
          int j_l = mo + r;
          float a = qv - k_s[j_l * C_ + ch] + pe[nt][r];
          st[j_l * C_ + ch] = (_Float16)a;
        }
      }
      WAVE_LDS_FENCE();
      v16h aa0 = load_a_frag(st, 0);
      v16h aa1 = load_a_frag(st, 32);

      // ---- attn MLP layer 1 (K=64), relu ----
      v8f t0 = bcast_bias(am_b1, 0);
      t0 = WMMA_F16(aa0, amw1[0][0], t0);
      t0 = relu8(WMMA_F16(aa1, amw1[1][0], t0));
      v8f t1 = bcast_bias(am_b1, 1);
      t1 = WMMA_F16(aa0, amw1[0][1], t1);
      t1 = relu8(WMMA_F16(aa1, amw1[1][1], t1));

      WAVE_LDS_FENCE();
      store_d_rows(t0, st, 0);
      store_d_rows(t1, st, 16);
      WAVE_LDS_FENCE();
      v16h at = load_a_frag(st, 0);

      // ---- attn MLP layer 2 -> logits; mask + online softmax + aggregate ----
#pragma unroll
      for (int nt = 0; nt < 4; ++nt) {
        v8f lg = WMMA_F16(at, amw2[nt], bcast_bias(am_b2, nt));
        int ch = nt * 16 + ch_lo;
#pragma unroll
        for (int r = 0; r < 8; ++r) {
          int j_l = mo + r;
          float l   = mask_s[j_l] ? lg[r] : NEG_;
          float val = v_s[j_l * C_ + ch] + pe[nt][r];
          float mold = mrun[ii][nt];
          float mnew = fmaxf(mold, l);
          float sc = __expf(mold - mnew);
          float w  = __expf(l - mnew);
          srun[ii][nt] = srun[ii][nt] * sc + w;
          arun[ii][nt] = arun[ii][nt] * sc + w * val;
          mrun[ii][nt] = mnew;
        }
      }
    }
  }

  // ---- combine lane halves (rows 0..7 live in lane, 8..15 in lane^16) ----
#pragma unroll
  for (int ii = 0; ii < 2; ++ii) {
    int i_loc = wave * 2 + ii;
#pragma unroll
    for (int nt = 0; nt < 4; ++nt) {
      float m1 = mrun[ii][nt], s1 = srun[ii][nt], a1 = arun[ii][nt];
      float m2 = swap16(m1), s2 = swap16(s1), a2 = swap16(a1);
      float M  = fmaxf(m1, m2);
      float e1 = __expf(m1 - M), e2 = __expf(m2 - M);
      float o  = (a1 * e1 + a2 * e2) / (s1 * e1 + s2 * e2);
      if (lane < 16) outpre_s[i_loc * C_ + nt * 16 + ch_lo] = o;
    }
  }
  __syncthreads();

  // ---- fused output linear: out = outpre @ out_w + out_b + x (wave 0) ----
  for (int idx = tid; idx < 16 * C_; idx += 256)
    stage[idx] = (_Float16)outpre_s[idx];
  __syncthreads();
  if (wave == 0) {
    v16h ax0 = load_a_frag(stage, 0);
    v16h ax1 = load_a_frag(stage, 32);
#pragma unroll
    for (int nt = 0; nt < 4; ++nt) {
      v16h w0 = load_b_frag(out_w, C_, 0, nt, C_);
      v16h w1 = load_b_frag(out_w, C_, 32, nt, C_);
      v8f o = bcast_bias(out_b, nt);
      o = WMMA_F16(ax0, w0, o);
      o = WMMA_F16(ax1, w1, o);
      int ch = nt * 16 + ch_lo;
#pragma unroll
      for (int r = 0; r < 8; ++r) {
        int gi = (b * N_ + ibase + mo + r) * C_ + ch;
        outp[gi] = o[r] + xg[gi];
      }
    }
  }
}

// ------------------------------- launcher ----------------------------------

extern "C" void kernel_launch(void* const* d_in, const int* in_sizes, int n_in,
                              void* d_out, int out_size, void* d_ws, size_t ws_size,
                              hipStream_t stream) {
  const float* coords = (const float*)d_in[0];
  const int*   maskp  = (const int*)d_in[1];
  const float* ce_w1  = (const float*)d_in[2];
  const float* ce_b1  = (const float*)d_in[3];
  const float* ce_w2  = (const float*)d_in[4];
  const float* ce_b2  = (const float*)d_in[5];
  const float* wq     = (const float*)d_in[6];
  const float* wk     = (const float*)d_in[7];
  const float* wv     = (const float*)d_in[8];
  const float* pm_w1  = (const float*)d_in[9];
  const float* pm_b1  = (const float*)d_in[10];
  const float* pm_w2  = (const float*)d_in[11];
  const float* pm_b2  = (const float*)d_in[12];
  const float* am_w1  = (const float*)d_in[13];
  const float* am_b1  = (const float*)d_in[14];
  const float* am_w2  = (const float*)d_in[15];
  const float* am_b2  = (const float*)d_in[16];
  const float* out_w  = (const float*)d_in[17];
  const float* out_b  = (const float*)d_in[18];
  float* outp = (float*)d_out;

  float* ws = (float*)d_ws;
  float* xg = ws + 0 * (size_t)BN_ * C_;
  float* qg = ws + 1 * (size_t)BN_ * C_;
  float* kg = ws + 2 * (size_t)BN_ * C_;
  float* vg = ws + 3 * (size_t)BN_ * C_;

  pt_embed_qkv<<<BN_, C_, 0, stream>>>(coords, ce_w1, ce_b1, ce_w2, ce_b2,
                                       wq, wk, wv, xg, qg, kg, vg);
  pt_attention<<<B_ * JTILES, 256, 0, stream>>>(
      coords, maskp, qg, kg, vg, xg,
      pm_w1, pm_b1, pm_w2, pm_b2, am_w1, am_b1, am_w2, am_b2,
      out_w, out_b, outp);
}
